// _rpn_47244640256457
// MI455X (gfx1250) — matrix operand order repaired and checked by reading, not compiled, not run on hardware
//
#include <hip/hip_runtime.h>
#include <hip/hip_bf16.h>

typedef __bf16 bf16;
typedef __bf16 bf16x8 __attribute__((ext_vector_type(8)));
typedef __bf16 v16bf  __attribute__((ext_vector_type(16)));
typedef float  v8f    __attribute__((ext_vector_type(8)));
typedef unsigned int u32x4 __attribute__((ext_vector_type(4)));
typedef int          i32x4 __attribute__((ext_vector_type(4)));
typedef int          i32x8 __attribute__((ext_vector_type(8)));

__device__ __forceinline__ v16bf frag_cat(bf16x8 lo, bf16x8 hi) {
    return __builtin_shufflevector(lo, hi, 0,1,2,3,4,5,6,7,8,9,10,11,12,13,14,15);
}

__device__ __forceinline__ v8f wmma_bf16(v16bf a, v16bf b, v8f c) {
    return __builtin_amdgcn_wmma_f32_16x16x32_bf16(false, a, false, b, (short)0, c, false, false);
}

// mish(x) = x * tanh(softplus(x)) = x * (t^2+2t)/(t^2+2t+2), t = e^x
__device__ __forceinline__ float mishf(float v) {
    if (v > 20.f) return v;
    float t = __expf(v);
    float u = t * t + 2.f * t;
    return v * (u / (u + 2.f));
}

// ---------------------------------------------------------------------------
// Tensor Data Mover: 1-D contiguous copy global -> LDS (D# built per ISA ch.8)
// ---------------------------------------------------------------------------
__device__ __forceinline__ void tdm_load_1d(const void* gptr, void* lptr, unsigned dwords) {
    unsigned lds = (unsigned)(size_t)lptr;                  // low 32b of generic = LDS byte addr
    unsigned long long ga = (unsigned long long)(size_t)gptr;
    u32x4 g0;
    g0[0] = 1u;                                             // count=1, user mode
    g0[1] = lds;                                            // lds_addr [63:32]
    g0[2] = (unsigned)(ga & 0xffffffffull);                 // global_addr lo
    g0[3] = (unsigned)((ga >> 32) & 0x01ffffffull) | (2u << 30); // ga hi, type=2 (image)
    i32x8 g1;
    g1[0] = 0x00020000;                                     // data_size = 4 bytes
    g1[1] = (int)(dwords << 16);                            // tensor_dim0[15:0] @ [31:16]
    g1[2] = (int)((dwords >> 16) | (1u << 16));             // tensor_dim0 hi | tensor_dim1=1
    g1[3] = (int)(dwords << 16);                            // tile_dim0 @ [31:16]
    g1[4] = 0;                                              // tile_dim1=0, tile_dim2=0
    g1[5] = (int)dwords;                                    // tensor_dim0_stride lo
    g1[6] = 0;
    g1[7] = 0;
    i32x4 z4 = {0, 0, 0, 0};
#if __clang_major__ >= 23
    i32x8 z8 = {0, 0, 0, 0, 0, 0, 0, 0};
    __builtin_amdgcn_tensor_load_to_lds(g0, g1, z4, z4, z8, 0);
#else
    __builtin_amdgcn_tensor_load_to_lds(g0, g1, z4, z4, 0);
#endif
}

// ---------------------------------------------------------------------------
// Kernel 1: repack conv3x3 weights OIHW(256,256,3,3) ->
//   wB3[(((kh*8+cb)*2+nb)*3+kw)*128 + oL][cL]   (bf16, 24KB contiguous per tile)
// ---------------------------------------------------------------------------
__global__ __launch_bounds__(256) void k_prep_wb(const float* __restrict__ Wb,
                                                 bf16* __restrict__ wB3) {
    int k = blockIdx.x;          // filt*256 + c
    int o = threadIdx.x;         // 0..255
    int filt = k >> 8;
    int c    = k & 255;
    int kh = filt / 3, kw = filt - kh * 3;
    int cb = c >> 5,   cL = c & 31;
    int nb = o >> 7,   oL = o & 127;
    size_t dst = ((((size_t)((kh * 8 + cb) * 2 + nb)) * 3 + kw) * 128 + oL) * 32 + cL;
    wB3[dst] = (bf16)Wb[(((size_t)o * 256 + c) * 3 + kh) * 3 + kw];
}

// ---------------------------------------------------------------------------
// Kernel 2: conv3x3 implicit GEMM + Mish, bf16 output (channels-last), stats.
// Grid: (1024 mblk = n*128+h, 2 nblk), 256 threads = 8 waves.
// Loop: 24 iterations over (kh, cblk); 3 kw handled by +-1 row offsets into
// an extended A row (w = -1..128). B tiles arrive via the Tensor Data Mover.
// ---------------------------------------------------------------------------
__global__ __launch_bounds__(256) void k_conv3x3(
        const float* __restrict__ x, const bf16* __restrict__ wB3,
        const float* __restrict__ bias, bf16* __restrict__ baseOut,
        float* __restrict__ bpartS, float* __restrict__ bpartQ) {
    __shared__ __align__(16) bf16 Alds[132 * 32];        // [w_ext=w+1][c]
    __shared__ __align__(16) bf16 Blds[3 * 128 * 32];    // [kw][o][c]
    __shared__ float pS[2][128], pQ[2][128];

    const int t    = threadIdx.x;
    const int lane = t & 31;
    const int wave = t >> 5;
    const int wm   = wave & 1;        // 2 waves over M
    const int wn   = wave >> 1;       // 4 waves over N
    const int mblk = blockIdx.x;      // n*128 + h
    const int nblk = blockIdx.y;
    const int n    = mblk >> 7;
    const int h    = mblk & 127;

    v8f acc[4][2];
    v8f z = {0.f,0.f,0.f,0.f,0.f,0.f,0.f,0.f};
#pragma unroll
    for (int i = 0; i < 4; ++i)
#pragma unroll
        for (int j = 0; j < 2; ++j) acc[i][j] = z;

    const int koffA = (lane < 16) ? 0 : 16;
    const int kbB   = (lane < 16) ? 0 : 16;

    // zero the always-out-of-bounds edge rows (w=-1 and w=128) once
    if (t < 32) {
        Alds[t]            = (bf16)0.f;     // row 0
        Alds[129 * 32 + t] = (bf16)0.f;     // row 129
    }

    for (int it = 0; it < 24; ++it) {
        const int kh = it >> 3;         // 0..2
        const int cb = it & 7;          // channel block
        const int hh = h + kh - 1;
        const bool hok = (hh >= 0) && (hh < 128);

        // --- issue TDM copy of this iteration's 24KB weight tile (wave 0) ---
        if (t < 32) {
            const bf16* gsrc = wB3 + ((size_t)((kh * 8 + cb) * 2 + nblk)) * (3 * 128 * 32);
            tdm_load_1d(gsrc, (void*)Blds, 6144);
        }

        // --- stage A: 128 rows x 32 channels, contiguous (no per-elem branch) ---
        {
            const int row = t >> 1;           // w position 0..127
            const int ch0 = (t & 1) << 4;     // 0 or 16
            __align__(16) bf16 tmp[16];
            if (hok) {
                const size_t base =
                    (((size_t)n * 256 + (cb << 5) + ch0) * 128 + hh) * 128 + row;
#pragma unroll
                for (int j = 0; j < 16; ++j)
                    tmp[j] = (bf16)x[base + (size_t)j * 16384];
            } else {
#pragma unroll
                for (int j = 0; j < 16; ++j) tmp[j] = (bf16)0.f;
            }
            bf16x8* dst = (bf16x8*)&Alds[(row + 1) * 32 + ch0];
            dst[0] = *(const bf16x8*)&tmp[0];
            dst[1] = *(const bf16x8*)&tmp[8];
        }
        if (t < 32) __builtin_amdgcn_s_wait_tensorcnt(0);
        __syncthreads();

        // --- compute: 3 kw shifts x (4 M-tiles x 2 N-tiles) WMMAs ---
#pragma unroll
        for (int kw = 0; kw < 3; ++kw) {
            v16bf bfrag[2];
#pragma unroll
            for (int tn = 0; tn < 2; ++tn) {
                int col = wn * 32 + tn * 16 + (lane & 15);
                const bf16* p = &Blds[(kw * 128 + col) * 32 + kbB];
                bfrag[tn] = frag_cat(*(const bf16x8*)p, *(const bf16x8*)(p + 8));
            }
#pragma unroll
            for (int tm = 0; tm < 4; ++tm) {
                int rowe = wm * 64 + tm * 16 + (lane & 15) + kw;   // w_ext = w + kw
                const bf16* p = &Alds[rowe * 32 + koffA];
                v16bf afrag = frag_cat(*(const bf16x8*)p, *(const bf16x8*)(p + 8));
#pragma unroll
                for (int tn = 0; tn < 2; ++tn)
                    acc[tm][tn] = wmma_bf16(afrag, bfrag[tn], acc[tm][tn]);
            }
        }
        __syncthreads();
    }

    // --- epilogue: bias + Mish, bf16 store, per-channel partial stats ---
#pragma unroll
    for (int tn = 0; tn < 2; ++tn) {
        const int colL = wn * 32 + tn * 16 + (lane & 15);
        const int colG = nblk * 128 + colL;
        const float bv = bias[colG];
        float s = 0.f, q = 0.f;
#pragma unroll
        for (int tm = 0; tm < 4; ++tm) {
#pragma unroll
            for (int r = 0; r < 8; ++r) {
                float v  = acc[tm][tn][r] + bv;
                float mv = mishf(v);
                s += mv; q += mv * mv;
                int row = wm * 64 + tm * 16 + r + ((lane < 16) ? 0 : 8);
                size_t m = (size_t)mblk * 128 + row;
                baseOut[m * 256 + colG] = (bf16)mv;
            }
        }
        s += __shfl_xor(s, 16, 32);
        q += __shfl_xor(q, 16, 32);
        if (lane < 16) { pS[wm][colL] = s; pQ[wm][colL] = q; }
    }
    __syncthreads();
    if (t < 128) {
        float S = pS[0][t] + pS[1][t];
        float Q = pQ[0][t] + pQ[1][t];
        bpartS[(size_t)(nblk * 128 + t) * 1024 + mblk] = S;
        bpartQ[(size_t)(nblk * 128 + t) * 1024 + mblk] = Q;
    }
}

// ---------------------------------------------------------------------------
// Kernel 3: deterministic reduce of base BN stats -> fold scale/shift
// ---------------------------------------------------------------------------
__global__ __launch_bounds__(256) void k_reduce_base(
        const float* __restrict__ bpartS, const float* __restrict__ bpartQ,
        const float* __restrict__ g, const float* __restrict__ be,
        float* __restrict__ sA, float* __restrict__ tA) {
    int c = threadIdx.x;  // 256
    float s = 0.f, q = 0.f;
    for (int i = 0; i < 1024; ++i) { s += bpartS[c * 1024 + i]; q += bpartQ[c * 1024 + i]; }
    const float inv = 1.f / 131072.f;
    float m  = s * inv;
    float v  = fmaxf(q * inv - m * m, 0.f);
    float rs = rsqrtf(v + 1e-5f);
    float sc = g[c] * rs;
    sA[c] = sc;
    tA[c] = be[c] - m * sc;
}

// ---------------------------------------------------------------------------
// Kernel 4: fold base-BN into 1x1 head weights (cls 0..17, reg 18..53, pad 64)
// ---------------------------------------------------------------------------
__global__ __launch_bounds__(256) void k_fold_head(
        const float* __restrict__ Wc, const float* __restrict__ bc,
        const float* __restrict__ Wr, const float* __restrict__ br,
        const float* __restrict__ sA, const float* __restrict__ tA,
        bf16* __restrict__ wH, float* __restrict__ biasH) {
    __shared__ float red[256];
    int o = blockIdx.x, c = threadIdx.x;
    float w = 0.f, b = 0.f;
    if (o < 18)      { w = Wc[o * 256 + c];        b = bc[o];      }
    else if (o < 54) { w = Wr[(o - 18) * 256 + c]; b = br[o - 18]; }
    wH[(size_t)o * 256 + c] = (bf16)(w * sA[c]);
    red[c] = w * tA[c];
    __syncthreads();
    for (int st = 128; st > 0; st >>= 1) {
        if (c < st) red[c] += red[c + st];
        __syncthreads();
    }
    if (c == 0) biasH[o] = b + red[0];
}

// ---------------------------------------------------------------------------
// Kernel 5: head GEMM (M=131072, N=64, K=256) via bf16 WMMA.
// ---------------------------------------------------------------------------
__global__ __launch_bounds__(256) void k_head_gemm(
        const bf16* __restrict__ baseOut, const bf16* __restrict__ wH,
        const float* __restrict__ biasH, float* __restrict__ hout,
        float* __restrict__ hpartS, float* __restrict__ hpartQ) {
    __shared__ __align__(16) bf16 Bl[64 * 256];   // [o][c]
    __shared__ float pS[8][64], pQ[8][64];

    const int t = threadIdx.x, lane = t & 31, wave = t >> 5;
    const int mblk = blockIdx.x;
    const int loff = (lane < 16) ? 0 : 16;
    const int boff = (lane < 16) ? 0 : 16;

    {
        const bf16x8* src = (const bf16x8*)wH;
        bf16x8* dst = (bf16x8*)Bl;
#pragma unroll
        for (int i = 0; i < 8; ++i) dst[t + i * 256] = src[t + i * 256];
    }
    __syncthreads();

    v8f acc[4];
    v8f z = {0.f,0.f,0.f,0.f,0.f,0.f,0.f,0.f};
#pragma unroll
    for (int i = 0; i < 4; ++i) acc[i] = z;

    const size_t mrow = (size_t)mblk * 128 + wave * 16 + (lane & 15);
#pragma unroll
    for (int kk = 0; kk < 8; ++kk) {
        const int k0 = kk * 32;
        const bf16* ap = baseOut + mrow * 256 + k0 + loff;
        v16bf afrag = frag_cat(*(const bf16x8*)ap, *(const bf16x8*)(ap + 8));
#pragma unroll
        for (int tn = 0; tn < 4; ++tn) {
            int col = tn * 16 + (lane & 15);
            const bf16* bp = &Bl[col * 256 + k0 + boff];
            v16bf bfrag = frag_cat(*(const bf16x8*)bp, *(const bf16x8*)(bp + 8));
            acc[tn] = wmma_bf16(afrag, bfrag, acc[tn]);
        }
    }

#pragma unroll
    for (int tn = 0; tn < 4; ++tn) {
        const int col = tn * 16 + (lane & 15);
        const float bv = biasH[col];
        float s = 0.f, q = 0.f;
#pragma unroll
        for (int r = 0; r < 8; ++r) {
            float v = acc[tn][r] + bv;
            int row = wave * 16 + r + ((lane < 16) ? 0 : 8);
            size_t m = (size_t)mblk * 128 + row;
            hout[m * 64 + col] = v;
            s += v; q += v * v;
        }
        s += __shfl_xor(s, 16, 32);
        q += __shfl_xor(q, 16, 32);
        if (lane < 16) { pS[wave][col] = s; pQ[wave][col] = q; }
    }
    __syncthreads();
    if (t < 64) {
        float S = 0.f, Q = 0.f;
#pragma unroll
        for (int w2 = 0; w2 < 8; ++w2) { S += pS[w2][t]; Q += pQ[w2][t]; }
        hpartS[(size_t)t * 1024 + mblk] = S;
        hpartQ[(size_t)t * 1024 + mblk] = Q;
    }
}

// ---------------------------------------------------------------------------
// Kernel 6: reduce head BN stats -> per-channel scale/shift
// ---------------------------------------------------------------------------
__global__ __launch_bounds__(64) void k_reduce_head(
        const float* __restrict__ hpartS, const float* __restrict__ hpartQ,
        const float* __restrict__ gC, const float* __restrict__ beC,
        const float* __restrict__ gR, const float* __restrict__ beR,
        float* __restrict__ hs, float* __restrict__ ht) {
    int o = threadIdx.x;  // 64
    float s = 0.f, q = 0.f;
    for (int i = 0; i < 1024; ++i) { s += hpartS[o * 1024 + i]; q += hpartQ[o * 1024 + i]; }
    const float inv = 1.f / 131072.f;
    float m  = s * inv;
    float v  = fmaxf(q * inv - m * m, 0.f);
    float rs = rsqrtf(v + 1e-5f);
    float g = 0.f, be = 0.f;
    if (o < 18)      { g = gC[o];      be = beC[o];      }
    else if (o < 54) { g = gR[o - 18]; be = beR[o - 18]; }
    float sc = g * rs;
    hs[o] = sc;
    ht[o] = be - m * sc;
}

// ---------------------------------------------------------------------------
// Kernel 7: softmax, anchor decode, clip, center, box encode -> d_out
// ---------------------------------------------------------------------------
__global__ __launch_bounds__(256) void k_finalize(
        const float* __restrict__ hout, const float* __restrict__ hs,
        const float* __restrict__ ht, const int* __restrict__ imgp,
        float* __restrict__ out) {
    int idx = blockIdx.x * 256 + threadIdx.x;
    if (idx >= 1179648) return;
    int m = idx / 9, a = idx - m * 9;
    int n = m >> 14, hw = m & 16383;
    int h = hw >> 7, w = hw & 127;

    const float* hv = hout + (size_t)m * 64;

    int c0 = 2 * a;
    float s0 = hv[c0]     * hs[c0]     + ht[c0];
    float s1 = hv[c0 + 1] * hs[c0 + 1] + ht[c0 + 1];
    float fg = 1.f / (1.f + expf(s0 - s1));

    float r[4];
#pragma unroll
    for (int j = 0; j < 4; ++j) {
        int c = 18 + 4 * a + j;
        r[j] = hv[c] * hs[c] + ht[c];
    }

    // base anchor (py-faster-rcnn, base_size=40; rintf == np.round half-even)
    int ridx = a / 3, sidx = a - ridx * 3;
    float ratio = (ridx == 0) ? 0.5f : ((ridx == 1) ? 1.f : 2.f);
    float scale = (sidx == 0) ? 8.f  : ((sidx == 1) ? 16.f : 32.f);
    float wsr = rintf(sqrtf(1600.f / ratio));
    float hsr = rintf(wsr * ratio);
    float Wb = wsr * scale, Hb = hsr * scale;
    float cx0 = 19.5f + 16.f * (float)w;
    float cy0 = 19.5f + 16.f * (float)h;
    float ax1 = cx0 - 0.5f * (Wb - 1.f), ay1 = cy0 - 0.5f * (Hb - 1.f);
    float ax2 = cx0 + 0.5f * (Wb - 1.f), ay2 = cy0 + 0.5f * (Hb - 1.f);

    float lim = (float)(*imgp);
    float d1 = fminf(fmaxf(r[0] + ax1, 0.f), lim);
    float d2 = fminf(fmaxf(r[1] + ay1, 0.f), lim);
    float d3 = fminf(fmaxf(r[2] + ax2, 0.f), lim);
    float d4 = fminf(fmaxf(r[3] + ay2, 0.f), lim);

    float rw  = d3 - d1, rh = d4 - d2;
    float rcx = d1 + 0.5f * rw, rcy = d2 + 0.5f * rh;
    float aw  = ax2 - ax1, ah = ay2 - ay1;
    float acx = ax1 + 0.5f * aw, acy = ay1 + 0.5f * ah;

    float tx = (rcx - acx) / aw, ty = (rcy - acy) / ah;
    float tw = logf(fmaxf(rw / aw, 1e-30f));
    float th = logf(fmaxf(rh / ah, 1e-30f));

    const int K = 147456;
    size_t base = (size_t)n * K + (size_t)(hw * 9 + a);
    out[base] = fg;
    float* ts = out + 1179648;
    ts[base * 4 + 0] = tx; ts[base * 4 + 1] = ty;
    ts[base * 4 + 2] = tw; ts[base * 4 + 3] = th;
    float* ro = out + 1179648 + 4718592;
    ro[base * 4 + 0] = rcx; ro[base * 4 + 1] = rcy;
    ro[base * 4 + 2] = rw;  ro[base * 4 + 3] = rh;
}

// ---------------------------------------------------------------------------
extern "C" void kernel_launch(void* const* d_in, const int* in_sizes, int n_in,
                              void* d_out, int out_size, void* d_ws, size_t ws_size,
                              hipStream_t stream) {
    (void)in_sizes; (void)n_in; (void)out_size; (void)ws_size;
    const float* x       = (const float*)d_in[0];
    const float* W_base  = (const float*)d_in[1];
    const float* b_base  = (const float*)d_in[2];
    const float* g_base  = (const float*)d_in[3];
    const float* be_base = (const float*)d_in[4];
    const float* W_cls   = (const float*)d_in[5];
    const float* b_cls   = (const float*)d_in[6];
    const float* g_cls   = (const float*)d_in[7];
    const float* be_cls  = (const float*)d_in[8];
    const float* W_reg   = (const float*)d_in[9];
    const float* b_reg   = (const float*)d_in[10];
    const float* g_reg   = (const float*)d_in[11];
    const float* be_reg  = (const float*)d_in[12];
    const int*   imgp    = (const int*)d_in[13];

    char* ws = (char*)d_ws;
    size_t off = 0;
    auto take = [&](size_t bytes) -> char* {
        char* p = ws + off;
        off += (bytes + 255) & ~(size_t)255;
        return p;
    };
    bf16*  wB3     = (bf16*)take((size_t)2304 * 256 * 2);
    bf16*  baseOut = (bf16*)take((size_t)131072 * 256 * 2);
    float* bpartS  = (float*)take((size_t)256 * 1024 * 4);
    float* bpartQ  = (float*)take((size_t)256 * 1024 * 4);
    float* sA      = (float*)take(256 * 4);
    float* tA      = (float*)take(256 * 4);
    bf16*  wH      = (bf16*)take((size_t)64 * 256 * 2);
    float* biasH   = (float*)take(64 * 4);
    float* hout    = (float*)take((size_t)131072 * 64 * 4);
    float* hpartS  = (float*)take((size_t)64 * 1024 * 4);
    float* hpartQ  = (float*)take((size_t)64 * 1024 * 4);
    float* hsArr   = (float*)take(64 * 4);
    float* htArr   = (float*)take(64 * 4);

    k_prep_wb    <<<2304, 256, 0, stream>>>(W_base, wB3);
    k_conv3x3    <<<dim3(1024, 2), 256, 0, stream>>>(x, wB3, b_base, baseOut, bpartS, bpartQ);
    k_reduce_base<<<1, 256, 0, stream>>>(bpartS, bpartQ, g_base, be_base, sA, tA);
    k_fold_head  <<<64, 256, 0, stream>>>(W_cls, b_cls, W_reg, b_reg, sA, tA, wH, biasH);
    k_head_gemm  <<<1024, 256, 0, stream>>>(baseOut, wH, biasH, hout, hpartS, hpartQ);
    k_reduce_head<<<1, 64, 0, stream>>>(hpartS, hpartQ, g_cls, be_cls, g_reg, be_reg, hsArr, htArr);
    k_finalize   <<<4608, 256, 0, stream>>>(hout, hsArr, htArr, imgp, (float*)d_out);
}
